// scaled_dot_product_module_6536940224839
// MI455X (gfx1250) — matrix operand order repaired and checked by reading, not compiled, not run on hardware
//
#include <hip/hip_runtime.h>

typedef __attribute__((ext_vector_type(16))) _Float16 v16h;
typedef __attribute__((ext_vector_type(8)))  _Float16 v8h;
typedef __attribute__((ext_vector_type(4)))  _Float16 v4h;
typedef __attribute__((ext_vector_type(8)))  float    v8f;

#define H_  16
#define S_  4096
#define D_  128
#define BM  128        // query rows per block (8 waves x 16)
#define BN  64         // keys per iteration
#define KP  136        // K LDS row pitch in halfs (128 + 8 pad; 272B rows, 16B aligned)
#define VP  80         // V^T LDS row pitch in halfs (64 + 16 pad; 160B rows, 16B aligned)
#define PP  72         // P LDS row pitch in halfs (64 + 8 pad; 144B rows, 16B aligned)

union V16U { v16h v; v8h h[2]; _Float16 e[16]; };

__global__ __launch_bounds__(256) void fa2_causal_wmma(
    const float* __restrict__ Qg, const float* __restrict__ Kg,
    const float* __restrict__ Vg, float* __restrict__ Og)
{
  __shared__ _Float16 Klds[BN * KP];       // 17408 B : K tile, row-major f16
  __shared__ _Float16 Vlds[D_ * VP];       // 20480 B : V tile, transposed (dim-major) f16
  __shared__ _Float16 Plds[8 * 16 * PP];   // 18432 B : per-wave P scratch (C/D -> A transpose)

  const int tid  = threadIdx.x;
  const int wave = tid >> 5;
  const int lane = tid & 31;
  const int l15  = lane & 15;
  const int half = lane >> 4;            // 0: lanes 0-15, 1: lanes 16-31

  const int qblk = blockIdx.x;
  const int head = blockIdx.y;
  const int q0   = qblk * BM;
  const int qb   = q0 + wave * 16;       // this wave's first query row
  const long hb  = (long)head * S_ * D_;

  // ---- Q A-fragments (f16, pre-scaled by 1/sqrt(D)), kept in registers ----
  // A layout 16x32 f16: lane L<16 -> row L, K = kc*32 + {0..7, 16..23};
  //                     lane L+16 -> row L, K = kc*32 + {8..15, 24..31}
  v16h qfrag[4];
  {
    const float sc = 0.08838834764831845f; // 1/sqrt(128)
    const float* qrow = Qg + hb + (long)(qb + l15) * D_;
    #pragma unroll
    for (int kc = 0; kc < 4; ++kc) {
      const int c1 = kc * 32 + half * 8;
      #pragma unroll
      for (int e = 0; e < 8; ++e) {
        qfrag[kc][e]     = (_Float16)(qrow[c1 + e] * sc);
        qfrag[kc][8 + e] = (_Float16)(qrow[c1 + 16 + e] * sc);
      }
    }
  }

  v8f Oacc[8];                 // O accumulator: 8 d-tiles of 16x16 f32 (C/D layout)
  const v8f vzero = {};
  #pragma unroll
  for (int d = 0; d < 8; ++d) Oacc[d] = vzero;

  float m_i[8], l_i[8];        // row stats; entry r <-> q row (qb + r + half*8)
  #pragma unroll
  for (int r = 0; r < 8; ++r) { m_i[r] = -__builtin_inff(); l_i[r] = 0.0f; }

  _Float16* myP = Plds + wave * (16 * PP);
  const int ntiles = (q0 + BM) / BN;     // causal: only tiles up to the block diagonal

  for (int j = 0; j < ntiles; ++j) {
    const int kv0 = j * BN;
    __syncthreads();                     // previous tile's LDS reads complete

    // ---- cooperative stage: K row-major f16, V transposed f16 ----
    {
      const int r  = tid >> 2;           // key row 0..63
      const int c0 = (tid & 3) * 32;     // dim chunk
      const float* kg = Kg + hb + (long)(kv0 + r) * D_ + c0;
      const float* vg = Vg + hb + (long)(kv0 + r) * D_ + c0;
      #pragma unroll
      for (int i = 0; i < 32; i += 4) {
        float4 k4 = *(const float4*)(kg + i);
        v4h h4;
        h4[0] = (_Float16)k4.x; h4[1] = (_Float16)k4.y;
        h4[2] = (_Float16)k4.z; h4[3] = (_Float16)k4.w;
        *(v4h*)(Klds + r * KP + c0 + i) = h4;
        float4 v4 = *(const float4*)(vg + i);
        Vlds[(c0 + i + 0) * VP + r] = (_Float16)v4.x;
        Vlds[(c0 + i + 1) * VP + r] = (_Float16)v4.y;
        Vlds[(c0 + i + 2) * VP + r] = (_Float16)v4.z;
        Vlds[(c0 + i + 3) * VP + r] = (_Float16)v4.w;
      }
      // prefetch next KV tile into cache while this tile is being consumed
      if (j + 1 < ntiles) {
        __builtin_prefetch(kg + (long)BN * D_, 0, 3);   // global_prefetch_b8
        __builtin_prefetch(vg + (long)BN * D_, 0, 3);
      }
    }
    __syncthreads();

    // Wave-uniform: skip tiles entirely above this wave's diagonal (all masked).
    // kv0 and qb are scalar per wave, so EXEC stays all-ones for the WMMA path.
    if (kv0 > qb + 15) continue;

    // ---- S = Q K^T : 4 n-tiles x 4 k-chunks of v_wmma_f32_16x16x32_f16 ----
    // B layout 32x16 f16: lane L<16 -> col N=L, K={0..7,16..23} ; lane L+16 -> col N=L, K={8..15,24..31}
    v8f Sacc[4];
    #pragma unroll
    for (int n = 0; n < 4; ++n) {
      v8f acc = vzero;
      const int key = n * 16 + l15;
      #pragma unroll
      for (int kc = 0; kc < 4; ++kc) {
        V16U b;
        const _Float16* bp = Klds + key * KP + kc * 32 + half * 8;
        b.h[0] = *(const v8h*)(bp);
        b.h[1] = *(const v8h*)(bp + 16);
        acc = __builtin_amdgcn_wmma_f32_16x16x32_f16(false, qfrag[kc], false, b.v,
                                                     (short)0, acc, false, false);
      }
      Sacc[n] = acc;
    }

    // ---- causal mask, only when this tile straddles the wave's diagonal ----
    // No element is masked when max(key)=kv0+63 <= min(qrow)=qb (wave-uniform test).
    if (kv0 + BN - 1 > qb) {
      #pragma unroll
      for (int n = 0; n < 4; ++n) {
        const int key = kv0 + n * 16 + l15;
        #pragma unroll
        for (int r = 0; r < 8; ++r) {
          const int qr = qb + r + half * 8;
          if (key > qr) Sacc[n][r] = -__builtin_inff();
        }
      }
    }

    // ---- online softmax: row max/sum via 16-lane xor-shuffle reduction ----
    #pragma unroll
    for (int r = 0; r < 8; ++r) {
      float t = fmaxf(fmaxf(Sacc[0][r], Sacc[1][r]), fmaxf(Sacc[2][r], Sacc[3][r]));
      t = fmaxf(t, __shfl_xor(t, 1));
      t = fmaxf(t, __shfl_xor(t, 2));
      t = fmaxf(t, __shfl_xor(t, 4));
      t = fmaxf(t, __shfl_xor(t, 8));
      const float mn    = fmaxf(m_i[r], t);
      const float alpha = __expf(m_i[r] - mn);   // exp(-inf)=0 on first tile
      m_i[r] = mn;
      float s = 0.0f;
      #pragma unroll
      for (int n = 0; n < 4; ++n) {
        const float p = __expf(Sacc[n][r] - mn); // masked entries -> 0
        Sacc[n][r] = p;
        s += p;
      }
      s += __shfl_xor(s, 1);
      s += __shfl_xor(s, 2);
      s += __shfl_xor(s, 4);
      s += __shfl_xor(s, 8);
      l_i[r] = l_i[r] * alpha + s;
      #pragma unroll
      for (int d = 0; d < 8; ++d) Oacc[d][r] *= alpha;
    }

    // ---- P: C/D layout -> row-major f16 in private LDS (transpose staging) ----
    #pragma unroll
    for (int r = 0; r < 8; ++r)
      #pragma unroll
      for (int n = 0; n < 4; ++n)
        myP[(r + half * 8) * PP + n * 16 + l15] = (_Float16)Sacc[n][r];

    // ---- O += P V : 2 k-chunks x 8 d-tiles of v_wmma_f32_16x16x32_f16 ----
    #pragma unroll
    for (int kc = 0; kc < 2; ++kc) {
      V16U a;   // A-fragment of P (wave-internal DS ordering makes store->load safe)
      const _Float16* ap = myP + l15 * PP + kc * 32 + half * 8;
      a.h[0] = *(const v8h*)(ap);
      a.h[1] = *(const v8h*)(ap + 16);
      #pragma unroll
      for (int d = 0; d < 8; ++d) {
        V16U b; // B-fragment of V from transposed LDS: keys {8h..8h+7, 16+8h..16+8h+7} per lane
        const _Float16* bp = Vlds + (d * 16 + l15) * VP + kc * 32 + half * 8;
        b.h[0] = *(const v8h*)(bp);
        b.h[1] = *(const v8h*)(bp + 16);
        Oacc[d] = __builtin_amdgcn_wmma_f32_16x16x32_f16(false, a.v, false, b.v,
                                                         (short)0, Oacc[d], false, false);
      }
    }
  }

  // ---- normalize by 1/l and store fp32 output ----
  #pragma unroll
  for (int r = 0; r < 8; ++r) {
    const float inv = 1.0f / l_i[r];
    float* op = Og + hb + (long)(qb + r + half * 8) * D_ + l15;
    #pragma unroll
    for (int d = 0; d < 8; ++d) op[d * 16] = Oacc[d][r] * inv;
  }
}

extern "C" void kernel_launch(void* const* d_in, const int* in_sizes, int n_in,
                              void* d_out, int out_size, void* d_ws, size_t ws_size,
                              hipStream_t stream) {
  (void)in_sizes; (void)n_in; (void)d_ws; (void)ws_size; (void)out_size;
  const float* Q = (const float*)d_in[0];
  const float* K = (const float*)d_in[1];
  const float* V = (const float*)d_in[2];
  // d_in[3] is the boolean causal mask; causality is computed analytically.
  float* O = (float*)d_out;
  dim3 grid(S_ / BM, H_);
  fa2_causal_wmma<<<grid, 256, 0, stream>>>(Q, K, V, O);
}
